// Retrieval_Retro_1941325218121
// MI455X (gfx1250) — compile-verified, matrix operand order checked by reading, not run on hardware
//
#include <hip/hip_runtime.h>

// ---------------------------------------------------------------------------
// Retrieval-Retro forward pass for MI455X (gfx1250, wave32, WMMA).
// All dense GEMMs use v_wmma_f32_16x16x32_f16 (f16 A/B, f32 accum).
// Edge-message GEMM fuses: row-gather (h[src]) -> concat(edge_attr) -> GEMM
// -> ReLU -> scatter-add (global_atomic_add_f32 into agg[dst]).
// ---------------------------------------------------------------------------

typedef __attribute__((ext_vector_type(16))) _Float16 v16h;
typedef __attribute__((ext_vector_type(8)))  float    v8f;

namespace cfg {
constexpr int B      = 32;
constexpr int N      = 50000;
constexpr int E      = 400000;
constexpr int K      = 3;
constexpr int NS     = 100;
constexpr int ES     = 800;
constexpr int IN     = 200;
constexpr int BOND   = 64;
constexpr int HID    = 128;
constexpr int LAYERS = 3;
constexpr int FF     = 512;
constexpr int OUT    = 1394;
constexpr int G      = B * K;            // 96 subgraphs per set
constexpr int AE     = G * ES;           // 76800 edges per set
constexpr int NT     = N + 2 * G * NS;   // 69200 total nodes (main + add + add2)
constexpr int KPMAX  = 256;              // max padded K
constexpr int LDA    = KPMAX + 8;        // LDS row pitch (halves)
constexpr int MROWS  = 64;               // rows per GEMM block
}

// ---- WMMA fragment loader --------------------------------------------------
// 16-bit A/B fragment layout (wave32): lanes 0-15 -> rows 0-15, K = k0+{0..7},
// k0+{16..23}; lanes 16-31 -> rows 0-15, K = k0+{8..15}, k0+{24..31}.
// Works for A (LDS, row-major over K) and for B via K-major (transposed) W.
__device__ __forceinline__ v16h rr_frag(const _Float16* __restrict__ base,
                                        int ld, int k0, int lane) {
  const int r  = lane & 15;
  const int ks = k0 + ((lane >> 4) << 3);
  const _Float16* p = base + r * ld + ks;
  v16h a;
#pragma unroll
  for (int j = 0; j < 8; ++j) a[j] = p[j];
#pragma unroll
  for (int j = 0; j < 8; ++j) a[8 + j] = p[16 + j];
  return a;
}

// ---- Generic fused GEMM ----------------------------------------------------
// out_row = act( concat(srcA[gA[row]], srcB2[gB[row]]) @ W + bias ) [+res]
// stored directly or atomically scattered to out[scatter[row]].
// W is pre-converted f16, K-major: Wt[n * Kpad + k], zero-padded to Kpad.
__global__ __launch_bounds__(128)
void rr_gemm(const float* __restrict__ srcA, int ldA, int KA,
             const int* __restrict__ gA,
             const float* __restrict__ srcB2, int ldB2, int KB2,
             const int* __restrict__ gB,
             const _Float16* __restrict__ Wt, int Kpad,
             const float* __restrict__ bias,
             const float* __restrict__ residual,
             const float* __restrict__ preluA,
             float* __restrict__ out,
             const int* __restrict__ scatter,
             int M, int act) {
  __shared__ _Float16 As[cfg::MROWS][cfg::LDA];
  const int tid  = threadIdx.x;
  const int lane = tid & 31;
  const int wave = tid >> 5;
  const int row0 = blockIdx.x * cfg::MROWS;
  const int Ktot = KA + KB2;

  // Stage gathered + concatenated rows into LDS as f16 (zero-padded).
  for (int r = 0; r < cfg::MROWS; ++r) {
    const int row = row0 + r;
    const bool ok = row < M;
    const int ia = ok ? (gA ? gA[row] : row) : 0;
    const int ib = (ok && srcB2) ? (gB ? gB[row] : row) : 0;
    for (int k = tid; k < Kpad; k += 128) {
      float v = 0.f;
      if (ok) {
        if (k < KA)        v = srcA[(size_t)ia * ldA + k];
        else if (k < Ktot) v = srcB2[(size_t)ib * ldB2 + (k - KA)];
      }
      As[r][k] = (_Float16)v;
    }
  }
  __syncthreads();

  constexpr int MT = cfg::MROWS / 16;
  v8f zero = {};
  v8f acc[MT][2];
#pragma unroll
  for (int mt = 0; mt < MT; ++mt) { acc[mt][0] = zero; acc[mt][1] = zero; }

  const int nb = wave * 32;  // each wave owns two 16-col tiles
  for (int k0 = 0; k0 < Kpad; k0 += 32) {
    v16h b0 = rr_frag(Wt + (size_t)nb * Kpad, Kpad, k0, lane);
    v16h b1 = rr_frag(Wt + (size_t)(nb + 16) * Kpad, Kpad, k0, lane);
#pragma unroll
    for (int mt = 0; mt < MT; ++mt) {
      v16h a = rr_frag(&As[mt * 16][0], cfg::LDA, k0, lane);
      acc[mt][0] = __builtin_amdgcn_wmma_f32_16x16x32_f16(
          false, a, false, b0, (short)0, acc[mt][0], false, false);
      acc[mt][1] = __builtin_amdgcn_wmma_f32_16x16x32_f16(
          false, a, false, b1, (short)0, acc[mt][1], false, false);
    }
  }

  const float aval = (act == 2 && preluA) ? preluA[0] : 0.f;
#pragma unroll
  for (int mt = 0; mt < MT; ++mt) {
#pragma unroll
    for (int tI = 0; tI < 2; ++tI) {
      const int col = nb + 16 * tI + (lane & 15);
      const float bv = bias ? bias[col] : 0.f;
#pragma unroll
      for (int i = 0; i < 8; ++i) {
        const int r = mt * 16 + i + ((lane >> 4) << 3);
        const int row = row0 + r;
        if (row < M) {
          float v = acc[mt][tI][i] + bv;
          if (act == 1)      v = fmaxf(v, 0.f);
          else if (act == 2) v = (v >= 0.f) ? v : aval * v;
          if (residual) v += residual[(size_t)row * cfg::HID + col];
          if (scatter)
            atomicAdd(out + (size_t)scatter[row] * cfg::HID + col, v);
          else
            out[(size_t)row * cfg::HID + col] = v;
        }
      }
    }
  }
}

// ---- Weight convert: f32 [K][128] -> f16 K-major [128][Kpad], zero-padded --
__global__ void rr_wconv(const float* __restrict__ w, _Float16* __restrict__ wt,
                         int Kd, int Nd, int Kpad) {
  const int idx = blockIdx.x * blockDim.x + threadIdx.x;
  if (idx >= Nd * Kpad) return;
  const int n = idx / Kpad, k = idx - n * Kpad;
  wt[(size_t)n * Kpad + k] = (k < Kd) ? (_Float16)w[(size_t)k * Nd + n]
                                      : (_Float16)0.f;
}

// ---- Globalize per-subgraph edge indices -----------------------------------
__global__ void rr_edges(const int* __restrict__ aei, int nodeOff,
                         int* __restrict__ srcG, int* __restrict__ dstG) {
  const int idx = blockIdx.x * blockDim.x + threadIdx.x;
  if (idx >= cfg::AE) return;
  const int g = idx / cfg::ES, e = idx - g * cfg::ES;
  const int base = nodeOff + g * cfg::NS;
  srcG[idx] = base + aei[(size_t)g * 2 * cfg::ES + e];
  dstG[idx] = base + aei[(size_t)g * 2 * cfg::ES + cfg::ES + e];
}

__global__ void rr_repidx(int* __restrict__ r) {
  const int g = threadIdx.x;
  if (g < cfg::G) r[g] = g / cfg::K;
}

// ---- Weighted segment-sum pooling ------------------------------------------
__global__ void rr_pool(const float* __restrict__ h, const float* __restrict__ w,
                        const int* __restrict__ seg, int nsDiv,
                        float* __restrict__ out, int nNodes) {
  const int idx = blockIdx.x * blockDim.x + threadIdx.x;
  if (idx >= nNodes * cfg::HID) return;
  const int node = idx >> 7, c = idx & 127;
  const int s = seg ? seg[node] : node / nsDiv;
  atomicAdd(out + (size_t)s * cfg::HID + c,
            h[(size_t)node * cfg::HID + c] * w[node]);
}

__global__ void rr_copy(const float* __restrict__ a, float* __restrict__ b,
                        int n) {
  const int i = blockIdx.x * blockDim.x + threadIdx.x;
  if (i < n) b[i] = a[i];
}

// ---- Pre-LN transformer encoder (self or MulT-style cross) -----------------
struct EncP {
  const float *wq, *wk, *wv, *wo;
  const float *ln1g, *ln1b, *lnkg, *lnkb, *ln2g, *ln2b;
  const float *ff1, *ffb1, *ff2, *ffb2, *lnfg, *lnfb;
};

__device__ __forceinline__ float rr_bsum(float v, volatile float* red, int t) {
#pragma unroll
  for (int o = 16; o > 0; o >>= 1) v += __shfl_down(v, o, 32);
  __syncthreads();
  if ((t & 31) == 0) red[t >> 5] = v;
  __syncthreads();
  return red[0] + red[1] + red[2] + red[3];
}

__global__ __launch_bounds__(128)
void rr_encoder(float* __restrict__ x, const float* __restrict__ kv,
                EncP p, int nL, int sq, int sk) {
  constexpr int H = cfg::HID, F = cfg::FF;
  const int b = blockIdx.x, t = threadIdx.x;
  __shared__ float xb[3][H], hs[3][H], ss[3][H];
  __shared__ float Qm[3][H], Km[3][H], Vm[3][H], Om[3][H];
  __shared__ float att[3][4][3];
  __shared__ float ffh[3][F];
  __shared__ float red[4];

  for (int s = 0; s < sq; ++s) xb[s][t] = x[(size_t)(b * sq + s) * H + t];
  __syncthreads();

  for (int l = 0; l < nL; ++l) {
    const float* wq = p.wq + (size_t)l * H * H;
    const float* wk = p.wk + (size_t)l * H * H;
    const float* wv = p.wv + (size_t)l * H * H;
    const float* wo = p.wo + (size_t)l * H * H;
    // h = LN1(x)
    for (int s = 0; s < sq; ++s) {
      float v = xb[s][t];
      float mu = rr_bsum(v, red, t) * (1.f / H);
      float d = v - mu;
      float var = rr_bsum(d * d, red, t) * (1.f / H);
      hs[s][t] = d * rsqrtf(var + 1e-5f) * p.ln1g[l * H + t] + p.ln1b[l * H + t];
    }
    float (*src)[H] = hs;
    const int skk = kv ? sk : sq;
    if (kv) {  // src = LNk(kv), recomputed every layer
      for (int s = 0; s < sk; ++s) {
        float v = kv[(size_t)(b * sk + s) * H + t];
        float mu = rr_bsum(v, red, t) * (1.f / H);
        float d = v - mu;
        float var = rr_bsum(d * d, red, t) * (1.f / H);
        ss[s][t] = d * rsqrtf(var + 1e-5f) * p.lnkg[l * H + t] + p.lnkb[l * H + t];
      }
      src = ss;
    }
    __syncthreads();
    // projections
    for (int s = 0; s < sq; ++s) {
      float a = 0.f;
      for (int k2 = 0; k2 < H; ++k2) a += hs[s][k2] * wq[k2 * H + t];
      Qm[s][t] = a;
    }
    for (int s = 0; s < skk; ++s) {
      float a = 0.f, c = 0.f;
      for (int k2 = 0; k2 < H; ++k2) {
        const float sv = src[s][k2];
        a += sv * wk[k2 * H + t];
        c += sv * wv[k2 * H + t];
      }
      Km[s][t] = a;
      Vm[s][t] = c;
    }
    __syncthreads();
    // scores + softmax (per (query, head))
    if (t < sq * 4) {
      const int s = t >> 2, hh = t & 3;
      float sc[3], mx = -1e30f;
      for (int k2 = 0; k2 < skk; ++k2) {
        float a = 0.f;
        for (int d2 = 0; d2 < 32; ++d2)
          a += Qm[s][hh * 32 + d2] * Km[k2][hh * 32 + d2];
        sc[k2] = a * 0.17677669529663687f;  // 1/sqrt(32)
        mx = fmaxf(mx, sc[k2]);
      }
      float den = 0.f;
      for (int k2 = 0; k2 < skk; ++k2) { sc[k2] = expf(sc[k2] - mx); den += sc[k2]; }
      for (int k2 = 0; k2 < skk; ++k2) att[s][hh][k2] = sc[k2] / den;
    }
    __syncthreads();
    // O = att @ V
    for (int s = 0; s < sq; ++s) {
      const int hh = t >> 5, d2 = t & 31;
      float a = 0.f;
      for (int k2 = 0; k2 < skk; ++k2) a += att[s][hh][k2] * Vm[k2][hh * 32 + d2];
      Om[s][t] = a;
    }
    __syncthreads();
    // x += O @ Wo
    for (int s = 0; s < sq; ++s) {
      float a = 0.f;
      for (int k2 = 0; k2 < H; ++k2) a += Om[s][k2] * wo[k2 * H + t];
      xb[s][t] += a;
    }
    __syncthreads();
    // FF block
    for (int s = 0; s < sq; ++s) {
      float v = xb[s][t];
      float mu = rr_bsum(v, red, t) * (1.f / H);
      float d = v - mu;
      float var = rr_bsum(d * d, red, t) * (1.f / H);
      hs[s][t] = d * rsqrtf(var + 1e-5f) * p.ln2g[l * H + t] + p.ln2b[l * H + t];
    }
    __syncthreads();
    const float* ff1 = p.ff1 + (size_t)l * H * F;
    const float* ff2 = p.ff2 + (size_t)l * F * H;
    for (int s = 0; s < sq; ++s) {
#pragma unroll
      for (int j = 0; j < 4; ++j) {
        const int col = t + j * H;
        float a = p.ffb1[l * F + col];
        for (int k2 = 0; k2 < H; ++k2) a += hs[s][k2] * ff1[k2 * F + col];
        ffh[s][col] = fmaxf(a, 0.f);
      }
    }
    __syncthreads();
    for (int s = 0; s < sq; ++s) {
      float a = p.ffb2[l * H + t];
      for (int k2 = 0; k2 < F; ++k2) a += ffh[s][k2] * ff2[k2 * H + t];
      xb[s][t] += a;
    }
    __syncthreads();
  }
  // final LN
  for (int s = 0; s < sq; ++s) {
    float v = xb[s][t];
    float mu = rr_bsum(v, red, t) * (1.f / H);
    float d = v - mu;
    float var = rr_bsum(d * d, red, t) * (1.f / H);
    x[(size_t)(b * sq + s) * H + t] =
        d * rsqrtf(var + 1e-5f) * p.lnfg[t] + p.lnfb[t];
  }
}

// ---- Classifier ------------------------------------------------------------
__global__ void rr_cls_hidden(const float* __restrict__ me,
                              const float* __restrict__ c1,
                              const float* __restrict__ c2,
                              const float* __restrict__ w1,
                              const float* __restrict__ b1,
                              const float* __restrict__ aP,
                              float* __restrict__ hc) {
  const int idx = blockIdx.x * blockDim.x + threadIdx.x;
  if (idx >= cfg::B * 3 * cfg::HID) return;
  const int bb = idx / (3 * cfg::HID), j = idx - bb * 3 * cfg::HID;
  float acc = b1[j];
  for (int k = 0; k < cfg::HID; ++k)
    acc += me[bb * cfg::HID + k] * w1[(size_t)k * 384 + j];
  for (int k = 0; k < cfg::HID; ++k)
    acc += c1[bb * cfg::HID + k] * w1[(size_t)(cfg::HID + k) * 384 + j];
  for (int k = 0; k < cfg::HID; ++k)
    acc += c2[bb * cfg::HID + k] * w1[(size_t)(2 * cfg::HID + k) * 384 + j];
  const float a = aP[0];
  hc[idx] = (acc >= 0.f) ? acc : a * acc;
}

__global__ void rr_cls_out(const float* __restrict__ hc,
                           const float* __restrict__ w2,
                           const float* __restrict__ b2,
                           float* __restrict__ out) {
  const int idx = blockIdx.x * blockDim.x + threadIdx.x;
  if (idx >= cfg::B * cfg::OUT) return;
  const int bb = idx / cfg::OUT, o = idx - bb * cfg::OUT;
  float acc = b2[o];
  for (int k = 0; k < 3 * cfg::HID; ++k)
    acc += hc[bb * 384 + k] * w2[(size_t)k * cfg::OUT + o];
  out[idx] = 1.f / (1.f + expf(-acc));
}

// ---------------------------------------------------------------------------
extern "C" void kernel_launch(void* const* d_in, const int* in_sizes, int n_in,
                              void* d_out, int out_size, void* d_ws,
                              size_t ws_size, hipStream_t stream) {
  using namespace cfg;
  (void)in_sizes; (void)n_in; (void)out_size; (void)ws_size;

  // ---- inputs (setup_inputs() dict-insertion order; params flattened
  //      recursively in dict-insertion order) --------------------------------
  auto F32 = [&](int i) { return (const float*)d_in[i]; };
  auto I32 = [&](int i) { return (const int*)d_in[i]; };
  const float* x    = F32(0);
  const int*   ei   = I32(1);   // [2][E]
  const float* ea   = F32(2);
  const int*   bat  = I32(3);
  const float* fcw  = F32(4);
  const float* ax   = F32(5);
  const int*   aei  = I32(6);
  const float* aea  = F32(7);
  const float* afw  = F32(8);
  const float* a2x  = F32(9);
  const int*   a2ei = I32(10);
  const float* a2ea = F32(11);
  const float* a2fw = F32(12);
  // params: gnn{emb_w,emb_b,msg_w,msg_b,upd_w,upd_b}, fuse1{w,b,a},
  // fuse2{w,b,a}, sa1(16), ca1(16), sa2(16), ca2(16), cls{w1,b1,a,w2,b2}
  const float* emb_w = F32(13); const float* emb_b = F32(14);
  const float* msg_w = F32(15); const float* msg_b = F32(16);
  const float* upd_w = F32(17); const float* upd_b = F32(18);
  const float* f1w = F32(19); const float* f1b = F32(20); const float* f1a = F32(21);
  const float* f2w = F32(22); const float* f2b = F32(23); const float* f2a = F32(24);
  auto mkEnc = [&](int b0) {
    EncP e;
    e.wq = F32(b0 + 0);  e.wk = F32(b0 + 1);  e.wv = F32(b0 + 2);  e.wo = F32(b0 + 3);
    e.ln1g = F32(b0 + 4); e.ln1b = F32(b0 + 5);
    e.lnkg = F32(b0 + 6); e.lnkb = F32(b0 + 7);
    e.ln2g = F32(b0 + 8); e.ln2b = F32(b0 + 9);
    e.ff1 = F32(b0 + 10); e.ffb1 = F32(b0 + 11);
    e.ff2 = F32(b0 + 12); e.ffb2 = F32(b0 + 13);
    e.lnfg = F32(b0 + 14); e.lnfb = F32(b0 + 15);
    return e;
  };
  EncP sa1 = mkEnc(25), ca1 = mkEnc(41), sa2 = mkEnc(57), ca2 = mkEnc(73);
  const float* cw1 = F32(89); const float* cb1 = F32(90); const float* cla = F32(91);
  const float* cw2 = F32(92); const float* cb2 = F32(93);

  // ---- workspace layout ----------------------------------------------------
  char* ws = (char*)d_ws;
  size_t off = 0;
  auto take = [&](size_t bytes) {
    char* p = ws + off;
    off += (bytes + 255) & ~(size_t)255;
    return p;
  };
  float*    hbuf  = (float*)take((size_t)NT * HID * 4);
  float*    agg   = (float*)take((size_t)NT * HID * 4);
  _Float16* embT  = (_Float16*)take((size_t)HID * 224 * 2);
  _Float16* msgT  = (_Float16*)take((size_t)LAYERS * HID * 192 * 2);
  _Float16* updT  = (_Float16*)take((size_t)LAYERS * HID * 256 * 2);
  _Float16* fu1T  = (_Float16*)take((size_t)HID * 256 * 2);
  _Float16* fu2T  = (_Float16*)take((size_t)HID * 256 * 2);
  int*      src1  = (int*)take((size_t)AE * 4);
  int*      dst1  = (int*)take((size_t)AE * 4);
  int*      src2  = (int*)take((size_t)AE * 4);
  int*      dst2  = (int*)take((size_t)AE * 4);
  int*      repI  = (int*)take((size_t)G * 4);
  float*    memb  = (float*)take((size_t)B * HID * 4);
  float*    pool1 = (float*)take((size_t)G * HID * 4);
  float*    pool2 = (float*)take((size_t)G * HID * 4);
  float*    z1    = (float*)take((size_t)G * HID * 4);
  float*    z2    = (float*)take((size_t)G * HID * 4);
  float*    q1    = (float*)take((size_t)B * HID * 4);
  float*    q2    = (float*)take((size_t)B * HID * 4);
  float*    hc    = (float*)take((size_t)B * 3 * HID * 4);

  auto cdiv = [](int a, int b) { return (a + b - 1) / b; };

  // ---- 1) convert weights to f16 K-major -----------------------------------
  rr_wconv<<<cdiv(HID * 224, 256), 256, 0, stream>>>(emb_w, embT, IN, HID, 224);
  for (int l = 0; l < LAYERS; ++l) {
    rr_wconv<<<cdiv(HID * 192, 256), 256, 0, stream>>>(
        msg_w + (size_t)l * 192 * HID, msgT + (size_t)l * HID * 192, 192, HID, 192);
    rr_wconv<<<cdiv(HID * 256, 256), 256, 0, stream>>>(
        upd_w + (size_t)l * 256 * HID, updT + (size_t)l * HID * 256, 256, HID, 256);
  }
  rr_wconv<<<cdiv(HID * 256, 256), 256, 0, stream>>>(f1w, fu1T, 256, HID, 256);
  rr_wconv<<<cdiv(HID * 256, 256), 256, 0, stream>>>(f2w, fu2T, 256, HID, 256);

  // ---- 2) globalize subgraph edge lists ------------------------------------
  rr_edges<<<cdiv(AE, 256), 256, 0, stream>>>(aei, N, src1, dst1);
  rr_edges<<<cdiv(AE, 256), 256, 0, stream>>>(a2ei, N + G * NS, src2, dst2);
  rr_repidx<<<1, 128, 0, stream>>>(repI);

  // ---- 3) node embedding (main + add + add2 into one h buffer) -------------
  rr_gemm<<<cdiv(N, MROWS), 128, 0, stream>>>(
      x, IN, IN, nullptr, nullptr, 0, 0, nullptr, embT, 224, emb_b,
      nullptr, nullptr, hbuf, nullptr, N, 0);
  rr_gemm<<<cdiv(G * NS, MROWS), 128, 0, stream>>>(
      ax, IN, IN, nullptr, nullptr, 0, 0, nullptr, embT, 224, emb_b,
      nullptr, nullptr, hbuf + (size_t)N * HID, nullptr, G * NS, 0);
  rr_gemm<<<cdiv(G * NS, MROWS), 128, 0, stream>>>(
      a2x, IN, IN, nullptr, nullptr, 0, 0, nullptr, embT, 224, emb_b,
      nullptr, nullptr, hbuf + (size_t)(N + G * NS) * HID, nullptr, G * NS, 0);

  // ---- 4) message passing layers -------------------------------------------
  for (int l = 0; l < LAYERS; ++l) {
    const _Float16* mW = msgT + (size_t)l * HID * 192;
    const _Float16* uW = updT + (size_t)l * HID * 256;
    const float* mB = msg_b + l * HID;
    const float* uB = upd_b + l * HID;
    hipMemsetAsync(agg, 0, (size_t)NT * HID * 4, stream);
    // m = relu([h[src]|ea] @ W + b); agg[dst] += m   (fused scatter-atomic)
    rr_gemm<<<cdiv(E, MROWS), 128, 0, stream>>>(
        hbuf, HID, HID, ei, ea, BOND, BOND, nullptr, mW, 192, mB,
        nullptr, nullptr, agg, ei + E, E, 1);
    rr_gemm<<<cdiv(AE, MROWS), 128, 0, stream>>>(
        hbuf, HID, HID, src1, aea, BOND, BOND, nullptr, mW, 192, mB,
        nullptr, nullptr, agg, dst1, AE, 1);
    rr_gemm<<<cdiv(AE, MROWS), 128, 0, stream>>>(
        hbuf, HID, HID, src2, a2ea, BOND, BOND, nullptr, mW, 192, mB,
        nullptr, nullptr, agg, dst2, AE, 1);
    // h = h + relu([h|agg] @ W + b)
    rr_gemm<<<cdiv(NT, MROWS), 128, 0, stream>>>(
        hbuf, HID, HID, nullptr, agg, HID, HID, nullptr, uW, 256, uB,
        hbuf, nullptr, hbuf, nullptr, NT, 1);
  }

  // ---- 5) weighted pooling -------------------------------------------------
  hipMemsetAsync(memb, 0, (size_t)B * HID * 4, stream);
  hipMemsetAsync(pool1, 0, (size_t)G * HID * 4, stream);
  hipMemsetAsync(pool2, 0, (size_t)G * HID * 4, stream);
  rr_pool<<<cdiv(N * HID, 256), 256, 0, stream>>>(hbuf, fcw, bat, 1, memb, N);
  rr_pool<<<cdiv(G * NS * HID, 256), 256, 0, stream>>>(
      hbuf + (size_t)N * HID, afw, nullptr, NS, pool1, G * NS);
  rr_pool<<<cdiv(G * NS * HID, 256), 256, 0, stream>>>(
      hbuf + (size_t)(N + G * NS) * HID, a2fw, nullptr, NS, pool2, G * NS);

  // ---- 6) fuse: z = prelu([pool|rep(main)] @ W + b) ------------------------
  rr_gemm<<<cdiv(G, MROWS), 128, 0, stream>>>(
      pool1, HID, HID, nullptr, memb, HID, HID, repI, fu1T, 256, f1b,
      nullptr, f1a, z1, nullptr, G, 2);
  rr_gemm<<<cdiv(G, MROWS), 128, 0, stream>>>(
      pool2, HID, HID, nullptr, memb, HID, HID, repI, fu2T, 256, f2b,
      nullptr, f2a, z2, nullptr, G, 2);

  // ---- 7) self-attention encoders over z (seq = K = 3) ---------------------
  rr_encoder<<<B, 128, 0, stream>>>(z1, nullptr, sa1, 2, K, K);
  rr_encoder<<<B, 128, 0, stream>>>(z2, nullptr, sa2, 2, K, K);

  // ---- 8) cross-attention: q = main_emb attends to z -----------------------
  rr_copy<<<cdiv(B * HID, 256), 256, 0, stream>>>(memb, q1, B * HID);
  rr_copy<<<cdiv(B * HID, 256), 256, 0, stream>>>(memb, q2, B * HID);
  rr_encoder<<<B, 128, 0, stream>>>(q1, z1, ca1, 2, 1, K);
  rr_encoder<<<B, 128, 0, stream>>>(q2, z2, ca2, 2, 1, K);

  // ---- 9) classifier -------------------------------------------------------
  rr_cls_hidden<<<cdiv(B * 3 * HID, 256), 256, 0, stream>>>(
      memb, q1, q2, cw1, cb1, cla, hc);
  rr_cls_out<<<cdiv(B * OUT, 256), 256, 0, stream>>>(hc, cw2, cb2,
                                                     (float*)d_out);
}
